// MSA_5196910428424
// MI455X (gfx1250) — matrix operand order repaired and checked by reading, not compiled, not run on hardware
//
#include <hip/hip_runtime.h>

#define BB 4
#define SS 2048
#define HH 12
#define DH 64
#define DD 768

typedef __attribute__((ext_vector_type(16))) _Float16 v16h;
typedef __attribute__((ext_vector_type(8)))  float    v8f;
typedef __attribute__((ext_vector_type(4)))  unsigned int u32x4;
typedef __attribute__((ext_vector_type(2)))  _Float16 h2;

union Frag { v16h h; u32x4 q[2]; };
union HU   { h2 h; int i; };

// Q pre-scale: 1/sqrt(64) * log2(e) so QK^T scores are already in base-2
// units; softmax becomes pure exp2 with no per-element multiply.
#define QSCALE (0.125f * 1.44269504f)

// WGP-scope prefetch (scope 0 -> pull into all cache levels on miss).
// __builtin_prefetch lowers to SCOPE_SYS which skips the WGP caches.
__device__ __forceinline__ void prefetch_wgp(const void* p) {
    asm volatile("global_prefetch_b8 %0, off" :: "v"(p) : "memory");
}

// XOR-butterfly step on a packed half2 via ds_swizzle_b32 (immediate lane
// pattern: new_lane = (lane & 0x1f) ^ MSK) — no index VGPRs, no clamp VALU.
template <int MSK>
__device__ __forceinline__ h2 pkmax_swz(h2 a) {
    HU u; u.h = a;
    HU v; v.i = __builtin_amdgcn_ds_swizzle(u.i, (MSK << 10) | 0x1f);
    u.h = __builtin_elementwise_max(u.h, v.h);
    return u.h;
}

// ---------------------------------------------------------------------------
// QKV projection: y[b,s,h,e] = sum_d x[b,s,h,d] * W[h][e][d] + bias[h][e]
// Q,K stored fp16 [B,H,S,64]; V stored TRANSPOSED fp16 [B,H,64,S] so the
// P@V WMMA B-fragment is a contiguous per-lane load.
// ---------------------------------------------------------------------------
__global__ __launch_bounds__(64) void qkv_proj_kernel(
    const float* __restrict__ x,
    const float* __restrict__ Wq, const float* __restrict__ Wk, const float* __restrict__ Wv,
    const float* __restrict__ bq, const float* __restrict__ bk, const float* __restrict__ bv,
    _Float16* __restrict__ Qg, _Float16* __restrict__ Kg, _Float16* __restrict__ Vt)
{
    const int sb4 = blockIdx.x;            // (b, s/4) flattened
    const int h   = blockIdx.y;
    const int e   = threadIdx.x;           // 0..63 output channel
    const int b   = sb4 / (SS / 4);
    const int s0  = (sb4 % (SS / 4)) * 4;  // 4 sequence rows per block: W reuse

    __shared__ float xs[4][DH];
#pragma unroll
    for (int i = 0; i < 4; ++i)
        xs[i][e] = x[((size_t)(b * SS + s0 + i)) * DD + h * DH + e];
    __syncthreads();

    const float* wq = Wq + ((size_t)h * DH + e) * DH;
    const float* wk = Wk + ((size_t)h * DH + e) * DH;
    const float* wv = Wv + ((size_t)h * DH + e) * DH;
    const float bqv = bq[h * DH + e], bkv = bk[h * DH + e], bvv = bv[h * DH + e];
    float aq[4], ak[4], av[4];
#pragma unroll
    for (int i = 0; i < 4; ++i) { aq[i] = bqv; ak[i] = bkv; av[i] = bvv; }

#pragma unroll 8
    for (int d = 0; d < DH; ++d) {
        const float wqv = wq[d], wkv = wk[d], wvv = wv[d];
#pragma unroll
        for (int i = 0; i < 4; ++i) {
            const float xv = xs[i][d];
            aq[i] = fmaf(xv, wqv, aq[i]);
            ak[i] = fmaf(xv, wkv, ak[i]);
            av[i] = fmaf(xv, wvv, av[i]);
        }
    }

    const size_t bh = (size_t)b * HH + h;
#pragma unroll
    for (int i = 0; i < 4; ++i) {
        const int s = s0 + i;
        Qg[(bh * SS + s) * DH + e] = (_Float16)(aq[i] * QSCALE);
        Kg[(bh * SS + s) * DH + e] = (_Float16)ak[i];
        Vt[(bh * DH + e) * SS + s] = (_Float16)av[i];             // transposed
    }
}

// ---------------------------------------------------------------------------
// Flash attention, one wave32 per (b, h, 16-row Q tile), 64-key blocks.
//   S(16x64) = Q @ K^T               -> 8x v_wmma_f32_16x16x32_f16
//   row-max: half2-packed butterfly  -> 16 ds_swizzle + packed f16 max
//   row-sum: l += P @ ones           -> 2x wmma
//   O += P @ V                       -> 8x wmma
// C/D layout: VGPR r, lane l -> M = r + 8*(l>>4), N = l&15.
// A f16 16x32: lane m=l&15, g=l>>4; half i<8 -> K=8g+i, i>=8 -> K=8+8g+i.
// B f16 32x16: lane n=l&15, g=l>>4; half i<8 -> K=8g+i, i>=8 -> K=8+8g+i.
// ---------------------------------------------------------------------------
__global__ __launch_bounds__(32) void flash_attn_kernel(
    const _Float16* __restrict__ Qg, const _Float16* __restrict__ Kg,
    const _Float16* __restrict__ Vt, float* __restrict__ out)
{
    const int tile = blockIdx.x;           // 16-row query tile
    const int h    = blockIdx.y;
    const int b    = blockIdx.z;
    const int l    = threadIdx.x;
    const int n    = l & 15;
    const int g    = l >> 4;
    const size_t bh = (size_t)b * HH + h;
    const _Float16* Qbh = Qg + bh * SS * DH;
    const _Float16* Kbh = Kg + bh * SS * DH;
    const _Float16* Vbh = Vt + bh * DH * (size_t)SS;
    const int row0 = tile * 16;

    __shared__ _Float16 Plds[16][64];      // C-layout -> A-layout bounce for P

    // Q A-fragments, d-chunks kk=0 (0..31), kk=1 (32..63); resident all loop.
    Frag aq[2];
#pragma unroll
    for (int kk = 0; kk < 2; ++kk) {
        const _Float16* p = Qbh + ((size_t)(row0 + n)) * DH + kk * 32 + 8 * g;
        aq[kk].q[0] = *(const u32x4*)p;
        aq[kk].q[1] = *(const u32x4*)(p + 16);
    }

    // all-ones B fragment: row-sum of P via WMMA (every output col == row sum)
    Frag one;
#pragma unroll
    for (int i = 0; i < 16; ++i) one.h[i] = (_Float16)1.0f;

    v8f o[4] = {};                 // O accumulator, 16x64 as 4 C tiles
    float mrow[8], lrow[8];
#pragma unroll
    for (int r = 0; r < 8; ++r) { mrow[r] = -1e30f; lrow[r] = 0.0f; }

    for (int j = 0; j < SS / 64; ++j) {
        const int key0 = j * 64;

        // WGP-scope prefetch of next 64-key K/V block while WMMAs run.
        if (j + 1 < SS / 64) {
            const _Float16* kn = Kbh + (size_t)(key0 + 64 + l) * DH;
            prefetch_wgp(kn);
            prefetch_wgp(kn + 32 * DH);
            const _Float16* vn = Vbh + (size_t)l * SS + key0 + 64;
            prefetch_wgp(vn);
            prefetch_wgp(vn + 32 * (size_t)SS);
        }

        // S tiles: t covers keys 16t..16t+15; K-dim (d=64) split in two wmmas
        v8f s[4];
#pragma unroll
        for (int t = 0; t < 4; ++t) {
            Frag kb0, kb1;
            const _Float16* p0 = Kbh + ((size_t)(key0 + 16 * t + n)) * DH + 8 * g;
            kb0.q[0] = *(const u32x4*)p0;
            kb0.q[1] = *(const u32x4*)(p0 + 16);
            const _Float16* p1 = p0 + 32;
            kb1.q[0] = *(const u32x4*)p1;
            kb1.q[1] = *(const u32x4*)(p1 + 16);
            v8f acc = {};
            acc = __builtin_amdgcn_wmma_f32_16x16x32_f16(false, aq[0].h, false, kb0.h, (short)0, acc, false, false);
            acc = __builtin_amdgcn_wmma_f32_16x16x32_f16(false, aq[1].h, false, kb1.h, (short)0, acc, false, false);
            s[t] = acc;
        }

        // Per-lane partial row maxima over the 4 score tiles.
        float rmf[8];
#pragma unroll
        for (int r = 0; r < 8; ++r)
            rmf[r] = fmaxf(fmaxf(s[0][r], s[1][r]), fmaxf(s[2][r], s[3][r]));

        // Cross-lane row max: two rows packed per 32-bit half2, XOR butterfly
        // via ds_swizzle (immediate pattern). m only needs to be ~max (softmax
        // is m-invariant), so f16 rounding is safe.
        float mnew[8];
#pragma unroll
        for (int p = 0; p < 4; ++p) {
            HU u;
            u.h[0] = (_Float16)rmf[2 * p];
            u.h[1] = (_Float16)rmf[2 * p + 1];
            u.h = pkmax_swz<1>(u.h);
            u.h = pkmax_swz<2>(u.h);
            u.h = pkmax_swz<4>(u.h);
            u.h = pkmax_swz<8>(u.h);
            mnew[2 * p]     = fmaxf(mrow[2 * p],     (float)u.h[0]);
            mnew[2 * p + 1] = fmaxf(mrow[2 * p + 1], (float)u.h[1]);
        }

        // Online rescale + exponentiate (scores already in log2 units).
#pragma unroll
        for (int r = 0; r < 8; ++r) {
            const float alpha = exp2f(mrow[r] - mnew[r]);
            mrow[r] = mnew[r];
            lrow[r] *= alpha;                       // row sum added via wmma below
#pragma unroll
            for (int c = 0; c < 4; ++c) o[c][r] *= alpha;
#pragma unroll
            for (int t = 0; t < 4; ++t) {
                const float pv = exp2f(s[t][r] - mnew[r]);
                Plds[r + 8 * g][16 * t + n] = (_Float16)pv;   // C-layout scatter
            }
        }

        // Re-gather P as two 16x32 A-fragments (same-wave LDS ops are in order).
        Frag pa[2];
#pragma unroll
        for (int q = 0; q < 2; ++q) {
            const _Float16* pp = &Plds[n][32 * q + 8 * g];
            pa[q].q[0] = *(const u32x4*)pp;
            pa[q].q[1] = *(const u32x4*)(pp + 16);
        }

        // Row sums of P via WMMA against the all-ones B matrix.
        {
            v8f rs = {};
            rs = __builtin_amdgcn_wmma_f32_16x16x32_f16(false, pa[0].h, false, one.h, (short)0, rs, false, false);
            rs = __builtin_amdgcn_wmma_f32_16x16x32_f16(false, pa[1].h, false, one.h, (short)0, rs, false, false);
#pragma unroll
            for (int r = 0; r < 8; ++r) lrow[r] += rs[r];
        }

        // O += P @ V over 4 output-column chunks; V^T gives contiguous B-frags.
#pragma unroll
        for (int c = 0; c < 4; ++c) {
#pragma unroll
            for (int q = 0; q < 2; ++q) {
                Frag vb;
                const _Float16* vp = Vbh + ((size_t)(16 * c + n)) * SS + key0 + 32 * q + 8 * g;
                vb.q[0] = *(const u32x4*)vp;
                vb.q[1] = *(const u32x4*)(vp + 16);
                o[c] = __builtin_amdgcn_wmma_f32_16x16x32_f16(false, pa[q].h, false, vb.h, (short)0, o[c], false, false);
            }
        }
    }

    // Normalize and write out[b, s, h*64 + e] (fp32).
#pragma unroll
    for (int r = 0; r < 8; ++r) {
        const float inv = 1.0f / lrow[r];
        const int   srow = row0 + r + 8 * g;
        float* op = out + ((size_t)(b * SS + srow)) * DD + h * DH;
#pragma unroll
        for (int c = 0; c < 4; ++c) op[16 * c + n] = o[c][r] * inv;
    }
}

// ---------------------------------------------------------------------------
extern "C" void kernel_launch(void* const* d_in, const int* in_sizes, int n_in,
                              void* d_out, int out_size, void* d_ws, size_t ws_size,
                              hipStream_t stream) {
    (void)in_sizes; (void)n_in; (void)out_size; (void)ws_size;
    const float* x  = (const float*)d_in[0];
    const float* Wq = (const float*)d_in[1];
    const float* Wk = (const float*)d_in[2];
    const float* Wv = (const float*)d_in[3];
    const float* bq = (const float*)d_in[4];
    const float* bk = (const float*)d_in[5];
    const float* bv = (const float*)d_in[6];
    float* out = (float*)d_out;

    const size_t n_qkv = (size_t)BB * HH * SS * DH;   // 6.29M halfs each
    _Float16* Qg = (_Float16*)d_ws;
    _Float16* Kg = Qg + n_qkv;
    _Float16* Vt = Kg + n_qkv;                        // 37.7 MB of d_ws

    qkv_proj_kernel<<<dim3(BB * SS / 4, HH), 64, 0, stream>>>(
        x, Wq, Wk, Wv, bq, bk, bv, Qg, Kg, Vt);
    flash_attn_kernel<<<dim3(SS / 16, HH, BB), 32, 0, stream>>>(
        Qg, Kg, Vt, out);
}
